// NERLSTM_CRF_39419209843285
// MI455X (gfx1250) — matrix operand order repaired and checked by reading, not compiled, not run on hardware
//
#include <hip/hip_runtime.h>
#include <hip/hip_bf16.h>

// ---------------------------------------------------------------------------
// BiLSTM-CRF forward for MI455X (gfx1250, wave32, WMMA f32<-f16 16x16x32)
//
//   1) embed_gather : gather embedding rows, cast f32 -> f16   [L*B, 256]
//   2) cast_f16     : W_ih_{f,b}, W_hh_{f,b} f32 -> f16 ; W_out -> padded f16
//   3) inproj (WMMA): gx = embF16 @ W_ih^T + b   -> f32 [L*B, 1024] per dir
//   4) lstm  (WMMA) : sequential scan; 4 blocks/dir over batch, h double-
//                     buffered in LDS (1 barrier/step), c in regs, next-step
//                     gx prefetched into regs to hide load latency behind WMMA.
//   5) emissions    : WMMA GEMM vs zero-padded W_out (N=16, 9 valid)
//   6) viterbi      : max-plus DP per batch row + backtrace
// ---------------------------------------------------------------------------

#define LL   512
#define BB   64
#define EE   256
#define HD   256          // per-direction hidden
#define G4   1024         // 4*HD
#define NT   9
#define ROWS (LL * BB)    // 32768

typedef __attribute__((ext_vector_type(16))) _Float16 v16h;
typedef __attribute__((ext_vector_type(8)))  _Float16 v8h;
typedef __attribute__((ext_vector_type(8)))  float    v8f;

// Assemble a 16-half WMMA fragment from two 16-byte chunks.
__device__ __forceinline__ v16h frag16(const _Float16* p0, const _Float16* p1) {
    union { v16h v; v8h h[2]; } u;
    u.h[0] = *(const v8h*)p0;
    u.h[1] = *(const v8h*)p1;
    return u.v;
}

__device__ __forceinline__ v8f wmma_f16(v16h a, v16h b, v8f c) {
    // (neg_a, A, neg_b, B, c_mod, C, reuse_a, reuse_b)
    return __builtin_amdgcn_wmma_f32_16x16x32_f16(false, a, false, b,
                                                  (short)0, c, false, false);
}

__device__ __forceinline__ float sigmoidf_(float x) {
    return 1.0f / (1.0f + __expf(-x));
}

// ---------------------------------------------------------------------------
// 1) Embedding gather + cast.  emb[l][b][e] = embed[x[b][l]][e]   (time-first)
// ---------------------------------------------------------------------------
__global__ void embed_gather(const int* __restrict__ x,       // [B][L]
                             const float* __restrict__ embed, // [V][E]
                             _Float16* __restrict__ embF16)   // [L*B][E]
{
    const int row = blockIdx.x;            // l*B + b
    const int e   = threadIdx.x;           // 0..255
    const int l   = row / BB;
    const int b   = row - l * BB;
    const int tok = x[b * LL + l];
    embF16[(size_t)row * EE + e] = (_Float16)embed[(size_t)tok * EE + e];
}

// ---------------------------------------------------------------------------
// 2) Generic f32 -> f16 cast (weights) + zero-padded W_out cast
// ---------------------------------------------------------------------------
__global__ void cast_f16(const float* __restrict__ src, _Float16* __restrict__ dst, int n) {
    int i = blockIdx.x * blockDim.x + threadIdx.x;
    if (i < n) dst[i] = (_Float16)src[i];
}

__global__ void cast_wout_pad(const float* __restrict__ Wout,  // [9][512]
                              _Float16* __restrict__ dst)      // [16][512]
{
    int i = blockIdx.x * blockDim.x + threadIdx.x;             // 0..8191
    if (i >= 16 * 2 * HD) return;
    const int t = i >> 9;
    dst[i] = (t < NT) ? (_Float16)Wout[i] : (_Float16)0.0f;
}

// ---------------------------------------------------------------------------
// 3) Input projection GEMM: C[M=32768, N=1024] = A[M,256] * W^T + bias
//    Block = 8 waves; wave tile 32x64 (2 M-tiles x 4 N-tiles); block 128x128.
// ---------------------------------------------------------------------------
__global__ __launch_bounds__(256) void inproj_kernel(
    const _Float16* __restrict__ A,    // [ROWS][EE]
    const _Float16* __restrict__ Wih,  // [2][G4][EE]
    const float* __restrict__ bias_f,
    const float* __restrict__ bias_b,
    float* __restrict__ gx_f,          // [ROWS][G4]
    float* __restrict__ gx_b)
{
    const int dir = blockIdx.z;
    const _Float16* W  = Wih + (size_t)dir * G4 * EE;
    const float* bias  = dir ? bias_b : bias_f;
    float*       C     = dir ? gx_b   : gx_f;

    const int wave = threadIdx.x >> 5;
    const int lane = threadIdx.x & 31;
    const int lh   = lane & 15;        // in-tile row/col
    const int hi   = lane >> 4;        // half-wave select

    const int Mbase = blockIdx.x * 128 + (wave & 3) * 32;
    const int Nbase = blockIdx.y * 128 + (wave >> 2) * 64;

    v8f acc[2][4] = {};

    #pragma unroll
    for (int kc = 0; kc < 8; ++kc) {   // K = 256 in chunks of 32
        v16h a[2], bfr[4];
        #pragma unroll
        for (int mt = 0; mt < 2; ++mt) {
            const _Float16* p = A + (size_t)(Mbase + mt * 16 + lh) * EE + kc * 32 + hi * 8;
            a[mt] = frag16(p, p + 16);
        }
        #pragma unroll
        for (int nt = 0; nt < 4; ++nt) {
            const _Float16* p = W + (size_t)(Nbase + nt * 16 + lh) * EE + kc * 32 + hi * 8;
            bfr[nt] = frag16(p, p + 16);
        }
        #pragma unroll
        for (int mt = 0; mt < 2; ++mt)
            #pragma unroll
            for (int nt = 0; nt < 4; ++nt)
                acc[mt][nt] = wmma_f16(a[mt], bfr[nt], acc[mt][nt]);
    }

    #pragma unroll
    for (int mt = 0; mt < 2; ++mt)
        #pragma unroll
        for (int nt = 0; nt < 4; ++nt) {
            const int col = Nbase + nt * 16 + lh;
            const float bv = bias[col];
            #pragma unroll
            for (int r = 0; r < 8; ++r) {
                const int row = Mbase + mt * 16 + r + 8 * hi;   // C/D layout
                C[(size_t)row * G4 + col] = acc[mt][nt][r] + bv;
            }
        }
}

// ---------------------------------------------------------------------------
// 4) LSTM recurrence. grid = (4 batch-chunks, 2 dirs), block = 512 (16 waves).
//    Wave w owns gate columns [w*16, w*16+16); computes all four gate tiles
//    (i,f,g,o) for its slice so the pointwise update is wave-local.
//    h double-buffered in LDS (one barrier/step); c in registers; next-step
//    gx prefetched into registers to overlap global latency with WMMA.
// ---------------------------------------------------------------------------
__global__ __launch_bounds__(512) void lstm_kernel(
    const _Float16* __restrict__ whh16,  // [2][G4][HD] f16 row-major
    const float* __restrict__ gx_f,      // [L][B][G4]
    const float* __restrict__ gx_b,
    _Float16* __restrict__ hs)           // [L][B][512]  (fwd: 0..255, bwd: 256..511)
{
    const int dir   = blockIdx.y;
    const int bBase = blockIdx.x * 16;
    const int wave  = threadIdx.x >> 5;
    const int lane  = threadIdx.x & 31;
    const int lh    = lane & 15;
    const int hi    = lane >> 4;

    const _Float16* W  = whh16 + (size_t)dir * G4 * HD;
    const float*    gx = dir ? gx_b : gx_f;

    __shared__ _Float16 hlds[2][16][HD + 8];       // double-buffered, padded

    for (int i = threadIdx.x; i < 2 * 16 * (HD + 8); i += blockDim.x)
        (&hlds[0][0][0])[i] = (_Float16)0.0f;

    float c[8];
    #pragma unroll
    for (int r = 0; r < 8; ++r) c[r] = 0.0f;

    const int j = wave * 16 + lh;                  // hidden index this lane owns
    const _Float16* wb[4];                         // B-frag bases for i,f,g,o
    #pragma unroll
    for (int g = 0; g < 4; ++g)
        wb[g] = W + (size_t)(g * HD + j) * HD + hi * 8;

    // Prefetch gx for the first step.
    float gxc[4][8];
    {
        const int l0 = dir ? (LL - 1) : 0;
        const size_t base = ((size_t)l0 * BB + bBase) * G4;
        #pragma unroll
        for (int r = 0; r < 8; ++r) {
            const int bl = r + 8 * hi;
            const float* gp = gx + base + (size_t)bl * G4;
            gxc[0][r] = gp[j];
            gxc[1][r] = gp[HD + j];
            gxc[2][r] = gp[2 * HD + j];
            gxc[3][r] = gp[3 * HD + j];
        }
    }

    __syncthreads();

    for (int s = 0; s < LL; ++s) {
        const int l  = dir ? (LL - 1 - s) : s;
        const int rb = s & 1;                      // read buffer
        const int wbuf = rb ^ 1;                   // write buffer

        // ---- prefetch next step's gx (independent of everything below) ----
        float gxn[4][8];
        if (s + 1 < LL) {
            const int ln = dir ? (LL - 2 - s) : (s + 1);
            const size_t base = ((size_t)ln * BB + bBase) * G4;
            #pragma unroll
            for (int r = 0; r < 8; ++r) {
                const int bl = r + 8 * hi;
                const float* gp = gx + base + (size_t)bl * G4;
                gxn[0][r] = gp[j];
                gxn[1][r] = gp[HD + j];
                gxn[2][r] = gp[2 * HD + j];
                gxn[3][r] = gp[3 * HD + j];
            }
        }

        // ---- recurrent GEMM: gates = h @ W_hh^T ----
        v8f aI = {}, aF = {}, aG = {}, aO = {};
        #pragma unroll
        for (int kc = 0; kc < 8; ++kc) {
            const _Float16* ap = &hlds[rb][lh][kc * 32 + hi * 8];
            v16h a  = frag16(ap, ap + 16);
            v16h bi = frag16(wb[0] + kc * 32, wb[0] + kc * 32 + 16);
            v16h bf = frag16(wb[1] + kc * 32, wb[1] + kc * 32 + 16);
            v16h bg = frag16(wb[2] + kc * 32, wb[2] + kc * 32 + 16);
            v16h bo = frag16(wb[3] + kc * 32, wb[3] + kc * 32 + 16);
            aI = wmma_f16(a, bi, aI);
            aF = wmma_f16(a, bf, aF);
            aG = wmma_f16(a, bg, aG);
            aO = wmma_f16(a, bo, aO);
        }

        // ---- pointwise LSTM cell update ----
        float hval[8];
        #pragma unroll
        for (int r = 0; r < 8; ++r) {
            const float gi = aI[r] + gxc[0][r];
            const float gf = aF[r] + gxc[1][r];
            const float gg = aG[r] + gxc[2][r];
            const float go = aO[r] + gxc[3][r];
            c[r]    = sigmoidf_(gf) * c[r] + sigmoidf_(gi) * tanhf(gg);
            hval[r] = sigmoidf_(go) * tanhf(c[r]);
        }

        // ---- publish h (write buffer: no read hazard this step) ----
        #pragma unroll
        for (int r = 0; r < 8; ++r) {
            const int bl = r + 8 * hi;
            const _Float16 hh = (_Float16)hval[r];
            hlds[wbuf][bl][j] = hh;
            hs[((size_t)l * BB + bBase + bl) * (2 * HD) + dir * HD + j] = hh;
        }

        // rotate prefetched gx
        #pragma unroll
        for (int g = 0; g < 4; ++g)
            #pragma unroll
            for (int r = 0; r < 8; ++r)
                gxc[g][r] = gxn[g][r];

        __syncthreads();                           // h visible for next step
    }
}

// ---------------------------------------------------------------------------
// 5) Emissions (WMMA): em[row][t] = hs[row][:] . WoutPad[t][:] + b_out[t]
//    WoutPad is [16][512] f16 with rows 9..15 zeroed. Each wave -> one
//    16(M) x 16(N) tile over K = 512; lanes with col < 9 store.
// ---------------------------------------------------------------------------
__global__ __launch_bounds__(128) void emis_kernel(
    const _Float16* __restrict__ hsmat,    // [ROWS][512]
    const _Float16* __restrict__ WoutPad,  // [16][512]
    const float* __restrict__ bout,        // [9]
    float* __restrict__ em)                // [ROWS][9]
{
    const int wave = threadIdx.x >> 5;
    const int lane = threadIdx.x & 31;
    const int lh   = lane & 15;
    const int hi   = lane >> 4;
    const int Mbase = (blockIdx.x * 4 + wave) * 16;

    v8f acc = {};
    #pragma unroll
    for (int kc = 0; kc < 16; ++kc) {      // K = 512 in chunks of 32
        const _Float16* pa = hsmat + (size_t)(Mbase + lh) * (2 * HD) + kc * 32 + hi * 8;
        const _Float16* pb = WoutPad + (size_t)lh * (2 * HD) + kc * 32 + hi * 8;
        v16h a = frag16(pa, pa + 16);
        v16h b = frag16(pb, pb + 16);
        acc = wmma_f16(a, b, acc);
    }

    if (lh < NT) {
        const float bv = bout[lh];
        #pragma unroll
        for (int r = 0; r < 8; ++r) {
            const int row = Mbase + r + 8 * hi;
            em[(size_t)row * NT + lh] = acc[r] + bv;
        }
    }
}

// ---------------------------------------------------------------------------
// 6) Viterbi max-plus DP + backtrace. One thread per batch row.
//    out[0:32768] = tags [B][L] (as float), out[32768:32832] = best score.
// ---------------------------------------------------------------------------
__global__ void viterbi_kernel(const float* __restrict__ em,     // [L][B][9]
                               const float* __restrict__ start,
                               const float* __restrict__ trans,  // [9][9]
                               const float* __restrict__ endt,
                               unsigned char* __restrict__ hist, // [L][B][9]
                               float* __restrict__ out)
{
    __shared__ float s_tr[NT * NT];
    __shared__ float s_end[NT];
    for (int i = threadIdx.x; i < NT * NT; i += blockDim.x) s_tr[i] = trans[i];
    for (int i = threadIdx.x; i < NT; i += blockDim.x) s_end[i] = endt[i];
    __syncthreads();

    const int b = threadIdx.x;
    if (b >= BB) return;

    float score[NT];
    #pragma unroll
    for (int t = 0; t < NT; ++t) score[t] = start[t] + em[(size_t)b * NT + t];

    for (int l = 1; l < LL; ++l) {
        const float* e = em + ((size_t)l * BB + b) * NT;
        float ns[NT];
        #pragma unroll
        for (int jj = 0; jj < NT; ++jj) {
            float best = -3.4e38f; int bi = 0;
            #pragma unroll
            for (int ii = 0; ii < NT; ++ii) {
                const float v = score[ii] + s_tr[ii * NT + jj];
                if (v > best) { best = v; bi = ii; }   // first max wins (argmax)
            }
            ns[jj] = best + e[jj];
            hist[((size_t)l * BB + b) * NT + jj] = (unsigned char)bi;
        }
        #pragma unroll
        for (int t = 0; t < NT; ++t) score[t] = ns[t];
    }

    float best = -3.4e38f; int last = 0;
    #pragma unroll
    for (int t = 0; t < NT; ++t) {
        score[t] += s_end[t];
        if (score[t] > best) { best = score[t]; last = t; }
    }
    out[ROWS + b] = best;
    out[(size_t)b * LL + (LL - 1)] = (float)last;
    for (int l = LL - 1; l >= 1; --l) {
        last = hist[((size_t)l * BB + b) * NT + last];
        out[(size_t)b * LL + (l - 1)] = (float)last;
    }
}

// ---------------------------------------------------------------------------
// Host launch
// ---------------------------------------------------------------------------
extern "C" void kernel_launch(void* const* d_in, const int* in_sizes, int n_in,
                              void* d_out, int out_size, void* d_ws, size_t ws_size,
                              hipStream_t stream) {
    (void)in_sizes; (void)n_in; (void)out_size; (void)ws_size;

    const int*   x       = (const int*)  d_in[0];
    const float* embed   = (const float*)d_in[1];
    const float* W_ih_f  = (const float*)d_in[2];
    const float* W_hh_f  = (const float*)d_in[3];
    const float* b_f     = (const float*)d_in[4];
    const float* W_ih_b  = (const float*)d_in[5];
    const float* W_hh_b  = (const float*)d_in[6];
    const float* b_b     = (const float*)d_in[7];
    const float* W_out   = (const float*)d_in[8];
    const float* b_out   = (const float*)d_in[9];
    const float* start_t = (const float*)d_in[10];
    const float* trans   = (const float*)d_in[11];
    const float* end_t   = (const float*)d_in[12];
    float* out = (float*)d_out;

    // ---- workspace carve-up (256 B aligned) ----
    char* ws = (char*)d_ws;
    size_t off = 0;
    auto take = [&](size_t bytes) -> char* {
        char* p = ws + off;
        off += (bytes + 255) & ~(size_t)255;
        return p;
    };
    _Float16* embF16  = (_Float16*)take((size_t)ROWS * EE * 2);          // 16 MB
    _Float16* wih16   = (_Float16*)take((size_t)2 * G4 * EE * 2);        // 1 MB
    _Float16* whh16   = (_Float16*)take((size_t)2 * G4 * HD * 2);        // 1 MB
    _Float16* woutPad = (_Float16*)take((size_t)16 * 2 * HD * 2);        // 16 KB
    float*    gx_f    = (float*)   take((size_t)ROWS * G4 * 4);          // 134 MB
    float*    gx_b    = (float*)   take((size_t)ROWS * G4 * 4);          // 134 MB
    _Float16* hsbuf   = (_Float16*)take((size_t)ROWS * 2 * HD * 2);      // 33.5 MB
    float*    em      = (float*)   take((size_t)ROWS * NT * 4);          // 1.2 MB
    unsigned char* hist = (unsigned char*)take((size_t)ROWS * NT);       // 0.3 MB

    // 1) embedding gather + cast
    embed_gather<<<ROWS, EE, 0, stream>>>(x, embed, embF16);

    // 2) weight casts (row-major preserved)
    const int WN = G4 * EE;  // 262144
    cast_f16<<<(WN + 255) / 256, 256, 0, stream>>>(W_ih_f, wih16,          WN);
    cast_f16<<<(WN + 255) / 256, 256, 0, stream>>>(W_ih_b, wih16 + WN,     WN);
    cast_f16<<<(WN + 255) / 256, 256, 0, stream>>>(W_hh_f, whh16,          WN);
    cast_f16<<<(WN + 255) / 256, 256, 0, stream>>>(W_hh_b, whh16 + WN,     WN);
    cast_wout_pad<<<(16 * 2 * HD + 255) / 256, 256, 0, stream>>>(W_out, woutPad);

    // 3) input projections, both directions (WMMA)
    inproj_kernel<<<dim3(ROWS / 128, G4 / 128, 2), 256, 0, stream>>>(
        embF16, wih16, b_f, b_b, gx_f, gx_b);

    // 4) LSTM scan (WMMA), 4 batch-chunks x 2 directions
    lstm_kernel<<<dim3(4, 2), 512, 0, stream>>>(whh16, gx_f, gx_b, hsbuf);

    // 5) emissions (WMMA, padded W_out)
    emis_kernel<<<ROWS / 64, 128, 0, stream>>>(hsbuf, woutPad, b_out, em);

    // 6) viterbi decode
    viterbi_kernel<<<1, 128, 0, stream>>>(em, start_t, trans, end_t, hist, out);
}
